// Net2_48086453846024
// MI455X (gfx1250) — matrix operand order repaired and checked by reading, not compiled, not run on hardware
//
#include <hip/hip_runtime.h>
#include <hip/hip_bf16.h>

// ---------------------------------------------------------------------------
// 2-layer GCN: h1 = x@W1 ; agg1 = scatter_add(h1[src] -> dst) ; a = relu(agg1)
//              h2 = a@W2 ; out  = scatter_add(h2[src] -> dst)
//
// GEMMs: v_wmma_f32_16x16x32_bf16, wave32. One wave owns a 16-row M-stripe and
// all N/16 output tiles (A fragment reused from registers). Weights are staged
// once per workgroup into LDS as bf16, transposed [n][k], so each lane's B
// fragment is one contiguous 32B LDS read (ds_load_b128 x2).
// Edge aggregation: global_atomic_add_f32 (unsafeAtomicAdd), float4 gathers.
// ---------------------------------------------------------------------------

typedef __attribute__((ext_vector_type(16))) __bf16 v16bf;
typedef __attribute__((ext_vector_type(8)))  float  v8f;

__global__ __launch_bounds__(256) void fill_zero_kernel(float* __restrict__ p, long long n) {
    long long i = (long long)blockIdx.x * blockDim.x + threadIdx.x;
    if (i < n) p[i] = 0.0f;
}

// C[M x N] = A[M x K] * B[K x N]; fp32 in/out, bf16 WMMA compute.
// M multiple of 16 (50000 is), N multiple of 16, K multiple of 32.
template <int K, int N, bool RELU>
__global__ __launch_bounds__(256) void gemm_wmma_bf16(const float* __restrict__ A,
                                                      const float* __restrict__ B,
                                                      float* __restrict__ C, int M) {
    constexpr int TN = N / 16;

    // Stage weights to LDS: bf16, transposed so Bs[n][k] (k contiguous).
    __shared__ alignas(32) __bf16 Bs[N][K];
    for (int i = threadIdx.x; i < K * N; i += blockDim.x) {
        const int k = i / N;
        const int n = i % N;                 // B row-major [k][n] -> Bs[n][k]
        Bs[n][k] = (__bf16)B[i];
    }
    __syncthreads();

    const int lane = threadIdx.x & 31;
    const int wave = threadIdx.x >> 5;
    const int tm   = blockIdx.x * (blockDim.x >> 5) + wave;   // 16-row M-stripe
    if (tm >= (M >> 4)) return;              // whole-wave exit; EXEC all-ones below

    const int half = lane >> 4;              // 0: lanes 0-15, 1: lanes 16-31
    const int l15  = lane & 15;
    const long long m = (long long)tm * 16 + l15;   // A row owned by this lane

    v8f acc[TN] = {};
#pragma unroll
    for (int kc = 0; kc < K; kc += 32) {
        // A 16x32 bf16 fragment: per lane one contiguous 16-float
        // run: [kc + 16*half, +16), matching the B fragment's k order.
        v16bf a;
#pragma unroll
        for (int e = 0; e < 16; ++e) {
            const int ka = kc + half * 16 + e;
            float av = __builtin_nontemporal_load(&A[m * K + ka]);  // single-pass stream
            if (RELU) av = fmaxf(av, 0.0f);
            a[e] = (__bf16)av;
        }
        // Reuse the A fragment across all N-tiles of this stripe.
#pragma unroll
        for (int tn = 0; tn < TN; ++tn) {
            const int n = tn * 16 + l15;     // B/C column owned by this lane
            // B 32x16 fragment: lanes 0-15 hold K=kc..kc+15, lanes 16-31 the
            // next 16 -> one contiguous 32B LDS read per lane.
            const v16bf b = *(const v16bf*)&Bs[n][kc + half * 16];
            acc[tn] = __builtin_amdgcn_wmma_f32_16x16x32_bf16(
                /*neg_a=*/false, a, /*neg_b=*/false, b,
                /*c_mod=*/(short)0, acc[tn], /*reuse_a=*/false, /*reuse_b=*/false);
        }
    }

    // C/D f32 16x16 layout: VGPR r holds row (r + 8*half), column = lane&15.
#pragma unroll
    for (int tn = 0; tn < TN; ++tn) {
#pragma unroll
        for (int r = 0; r < 8; ++r)
            C[(long long)(tm * 16 + r + 8 * half) * N + tn * 16 + l15] = acc[tn][r];
    }
}

// out[dst[e]*C + c] += h[src[e]*C + c], one float4 quad per thread.
// h (12.8/6.4 MB) is L2-resident on a 192 MB L2, so the random gather mostly
// hits L2; atomics resolve at L2.
template <int LOGC>
__global__ __launch_bounds__(256) void scatter_add_kernel(const float* __restrict__ h,
                                                          const int* __restrict__ src,
                                                          const int* __restrict__ dst,
                                                          float* __restrict__ out, int E) {
    constexpr int LOGQ = LOGC - 2;           // float4 quads per row
    long long tid = (long long)blockIdx.x * blockDim.x + threadIdx.x;
    int e = (int)(tid >> LOGQ);
    int q = (int)(tid & ((1 << LOGQ) - 1));
    if (e >= E) return;
    const int s = src[e];
    const int d = dst[e];
    const float4 v = *(const float4*)(h + (((long long)s) << LOGC) + (q << 2));
    float* o = out + (((long long)d) << LOGC) + (q << 2);
    unsafeAtomicAdd(o + 0, v.x);             // global_atomic_add_f32 (no CAS loop)
    unsafeAtomicAdd(o + 1, v.y);
    unsafeAtomicAdd(o + 2, v.z);
    unsafeAtomicAdd(o + 3, v.w);
}

extern "C" void kernel_launch(void* const* d_in, const int* in_sizes, int n_in,
                              void* d_out, int out_size, void* d_ws, size_t ws_size,
                              hipStream_t stream) {
    const float* x   = (const float*)d_in[0];   // [M,128]
    const int*   ei  = (const int*)d_in[1];     // [2,E]
    const float* W1  = (const float*)d_in[2];   // [128,64]
    const float* W2  = (const float*)d_in[3];   // [64,32]
    float*       out = (float*)d_out;           // [M,32]

    const int M = in_sizes[0] / 128;            // 50000 (multiple of 16)
    const int E = in_sizes[1] / 2;              // 800000
    const int* src = ei;
    const int* dst = ei + E;

    // Workspace: [h1: M*64 f32][agg1: M*64 f32]; h2 reuses h1's slot (M*32).
    float* h1  = (float*)d_ws;
    float* agg = h1 + (size_t)M * 64;
    float* h2  = h1;

    const int stripes = M >> 4;                 // 3125 M-stripes
    const unsigned gemm_blocks = (unsigned)((stripes + 7) / 8);   // 8 waves/block

    // 1) agg1 = 0
    {
        long long n = (long long)M * 64;
        fill_zero_kernel<<<(unsigned)((n + 255) / 256), 256, 0, stream>>>(agg, n);
    }
    // 2) h1 = x @ W1
    gemm_wmma_bf16<128, 64, false><<<gemm_blocks, 256, 0, stream>>>(x, W1, h1, M);
    // 3) agg1[dst] += h1[src]
    {
        long long n = (long long)E * (64 / 4);
        scatter_add_kernel<6><<<(unsigned)((n + 255) / 256), 256, 0, stream>>>(h1, src, dst, agg, E);
    }
    // 4) out = 0
    {
        long long n = (long long)M * 32;
        fill_zero_kernel<<<(unsigned)((n + 255) / 256), 256, 0, stream>>>(out, n);
    }
    // 5) h2 = relu(agg1) @ W2
    gemm_wmma_bf16<64, 32, true><<<gemm_blocks, 256, 0, stream>>>(agg, W2, h2, M);
    // 6) out[dst] += h2[src]
    {
        long long n = (long long)E * (32 / 4);
        scatter_add_kernel<5><<<(unsigned)((n + 255) / 256), 256, 0, stream>>>(h2, src, dst, out, E);
    }
}
